// BLSTMModel_78305843741048
// MI455X (gfx1250) — compile-verified, matrix operand order checked by reading, not run on hardware
//
#include <hip/hip_runtime.h>

// Model dims
#define VV 32000
#define EE 512
#define HH 512
#define BB 8
#define TT 256
#define MROWS (TT * BB)       // 2048
#define G4H  (4 * HH)         // 2048

typedef __attribute__((ext_vector_type(16))) __bf16 v16bf;
typedef __attribute__((ext_vector_type(8)))  float  v8f;
typedef __attribute__((ext_vector_type(4)))  unsigned v4u;

union FragU { v16bf v; v4u q[2]; };

static __device__ __forceinline__ v16bf load_frag(const unsigned short* p0,
                                                  const unsigned short* p1) {
  FragU f;
  f.q[0] = *reinterpret_cast<const v4u*>(p0);
  f.q[1] = *reinterpret_cast<const v4u*>(p1);
  return f.v;
}

static __device__ __forceinline__ v8f bf_wmma(v16bf a, v16bf b, v8f c) {
  return __builtin_amdgcn_wmma_f32_16x16x32_bf16(false, a, false, b,
                                                 (short)0, c, false, false);
}

static __device__ __forceinline__ unsigned short f2bf(float x) {
  union { float f; unsigned u; } v; v.f = x;
  unsigned u = v.u;
  u += 0x7FFFu + ((u >> 16) & 1u);   // round-to-nearest-even
  return (unsigned short)(u >> 16);
}

static __device__ __forceinline__ float sigm(float x) {
  return 1.0f / (1.0f + __expf(-x));
}

// ---------------------------------------------------------------- converts
__global__ void cvt_bf16_kernel(const float* __restrict__ in,
                                unsigned short* __restrict__ out, int n) {
  int i = blockIdx.x * blockDim.x + threadIdx.x;
  if (i < n) out[i] = f2bf(in[i]);
}

// embed gather -> time-major [T,B,E] bf16
__global__ void embed_kernel(const int* __restrict__ x,
                             const float* __restrict__ embed,
                             unsigned short* __restrict__ out) {
  int idx = blockIdx.x * blockDim.x + threadIdx.x;
  if (idx >= TT * BB * EE) return;
  int e  = idx & (EE - 1);
  int tb = idx >> 9;          // EE == 512
  int b  = tb & (BB - 1);
  int t  = tb >> 3;           // BB == 8
  int tok = x[b * TT + t];
  out[idx] = f2bf(embed[(size_t)tok * EE + e]);
}

// comb[b*T+t][0:512]=fwd, [512:1024]=bwd  (bf16 -> bf16 copy)
__global__ void combine_kernel(const unsigned short* __restrict__ f1,
                               const unsigned short* __restrict__ b1,
                               unsigned short* __restrict__ comb) {
  int idx = blockIdx.x * blockDim.x + threadIdx.x;
  if (idx >= MROWS * HH) return;
  int n = idx & (HH - 1);
  int m = idx >> 9;           // m = b*T + t
  int b = m >> 8;             // TT == 256
  int t = m & (TT - 1);
  int src = (t * BB + b) * HH + n;
  comb[(size_t)m * (2 * HH) + n]      = f1[src];
  comb[(size_t)m * (2 * HH) + HH + n] = b1[src];
}

// ------------------------------------------------- WMMA GEMM: C = A @ W^T + bias
// A[M,K] bf16 row-major, W[N,K] bf16 row-major, C[M,N] f32.
// block (32,4): 4 waves; wave tile = 32x64 (2 A frags x 4 B frags = 8 WMMA/stage),
// K loop software-pipelined with ping-pong fragment buffers.
// grid (N/256, M/32)
struct GFrag { v16bf a0, a1, b0, b1, b2, b3; };

static __device__ __forceinline__ void g_load(const unsigned short* __restrict__ A,
                                              const unsigned short* __restrict__ W,
                                              size_t K, int m0, int n0,
                                              int ln, int half, int k0, GFrag& f) {
  const unsigned short* aP0 = A + (size_t)(m0 + ln) * K + k0 + half * 8;
  const unsigned short* aP1 = aP0 + (size_t)16 * K;
  f.a0 = load_frag(aP0, aP0 + 16);
  f.a1 = load_frag(aP1, aP1 + 16);
  const unsigned short* bP = W + (size_t)(n0 + ln) * K + k0 + half * 8;
  f.b0 = load_frag(bP, bP + 16);
  f.b1 = load_frag(bP + (size_t)16 * K, bP + (size_t)16 * K + 16);
  f.b2 = load_frag(bP + (size_t)32 * K, bP + (size_t)32 * K + 16);
  f.b3 = load_frag(bP + (size_t)48 * K, bP + (size_t)48 * K + 16);
}

static __device__ __forceinline__ void g_mma(const GFrag& f, v8f acc[8]) {
  acc[0] = bf_wmma(f.a0, f.b0, acc[0]);
  acc[1] = bf_wmma(f.a0, f.b1, acc[1]);
  acc[2] = bf_wmma(f.a0, f.b2, acc[2]);
  acc[3] = bf_wmma(f.a0, f.b3, acc[3]);
  acc[4] = bf_wmma(f.a1, f.b0, acc[4]);
  acc[5] = bf_wmma(f.a1, f.b1, acc[5]);
  acc[6] = bf_wmma(f.a1, f.b2, acc[6]);
  acc[7] = bf_wmma(f.a1, f.b3, acc[7]);
}

__global__ __launch_bounds__(128)
void gemm_bias_kernel(const unsigned short* __restrict__ A,
                      const unsigned short* __restrict__ W,
                      const float* __restrict__ bias,
                      float* __restrict__ C,
                      int M, int N, int K) {
  const int lane = threadIdx.x;
  const int wave = threadIdx.y;
  const int half = lane >> 4;
  const int ln   = lane & 15;
  const int m0 = blockIdx.y * 32;
  const int n0 = (blockIdx.x * 4 + wave) * 64;

  v8f acc[8];
  const v8f zero = {0.f, 0.f, 0.f, 0.f, 0.f, 0.f, 0.f, 0.f};
#pragma unroll
  for (int j = 0; j < 8; ++j) acc[j] = zero;

  GFrag f0, f1;
  g_load(A, W, (size_t)K, m0, n0, ln, half, 0, f0);
  // K is a multiple of 64 for all call sites (512 / 1024)
  for (int k0 = 0; k0 < K; k0 += 64) {
    g_load(A, W, (size_t)K, m0, n0, ln, half, k0 + 32, f1);
    g_mma(f0, acc);
    int kn = (k0 + 64 < K) ? (k0 + 64) : 0;   // final prefetch is a dummy
    g_load(A, W, (size_t)K, m0, n0, ln, half, kn, f0);
    g_mma(f1, acc);
  }

#pragma unroll
  for (int r = 0; r < 2; ++r) {
#pragma unroll
    for (int j = 0; j < 4; ++j) {
      int cN = n0 + j * 16 + ln;
      float bv = bias[cN];
#pragma unroll
      for (int i = 0; i < 8; ++i) {
        int row = m0 + r * 16 + half * 8 + i;
        C[(size_t)row * N + cN] = acc[r * 4 + j][i] + bv;
      }
    }
  }
}

// ------------------------------------------------- LSTM recurrence
// grid = 2 blocks (0 = fwd, 1 = bwd reversed), block (32,32) = 32 waves.
// Wave w owns hidden columns [w*16, w*16+16); computes i/f/g/o tiles itself.
// h state (bf16) in LDS (rows 8..15 stay zero to pad batch to 16); c in regs.
// K loop ping-pong pipelined: A frags via ds_load_b128, B frags (Whh) from L2.
struct LFrag { v16bf a, b0, b1, b2, b3; };

static __device__ __forceinline__ void l_load(const unsigned short* h_lds,
                                              const unsigned short* __restrict__ whh,
                                              int ln, int half, int col, int k0,
                                              LFrag& f) {
  const unsigned short* aP = h_lds + ln * HH + k0 + half * 8;
  f.a = load_frag(aP, aP + 16);
  const unsigned short* bP = whh + (size_t)col * HH + k0 + half * 8;
  const size_t gs = (size_t)HH * HH;          // gate stride = 512 rows
  f.b0 = load_frag(bP, bP + 16);
  f.b1 = load_frag(bP + gs, bP + gs + 16);
  f.b2 = load_frag(bP + 2 * gs, bP + 2 * gs + 16);
  f.b3 = load_frag(bP + 3 * gs, bP + 3 * gs + 16);
}

static __device__ __forceinline__ void l_mma(const LFrag& f, v8f acc[4]) {
  acc[0] = bf_wmma(f.a, f.b0, acc[0]);
  acc[1] = bf_wmma(f.a, f.b1, acc[1]);
  acc[2] = bf_wmma(f.a, f.b2, acc[2]);
  acc[3] = bf_wmma(f.a, f.b3, acc[3]);
}

__global__ __launch_bounds__(1024, 1)
void lstm_layer_kernel(const float* __restrict__ gates_fwd,
                       const float* __restrict__ gates_bwd,
                       const unsigned short* __restrict__ whh_fwd,
                       const unsigned short* __restrict__ whh_bwd,
                       unsigned short* __restrict__ out_fwd,
                       unsigned short* __restrict__ out_bwd,
                       float* __restrict__ hT_dst,   // fwd only
                       float* __restrict__ cT_dst) { // fwd only
  const int dir = blockIdx.x;
  const float*          gp  = dir ? gates_bwd : gates_fwd;
  const unsigned short* whh = dir ? whh_bwd   : whh_fwd;
  unsigned short*       out = dir ? out_bwd   : out_fwd;

  const int lane = threadIdx.x;
  const int wave = threadIdx.y;
  const int half = lane >> 4;
  const int ln   = lane & 15;
  const int col  = wave * 16 + ln;   // hidden column 0..511

  __shared__ __align__(16) unsigned short h_lds[16 * HH];
  for (int i = wave * 32 + lane; i < 16 * HH; i += 1024) h_lds[i] = 0;
  __syncthreads();

  float c_reg[8], h_reg[8];
#pragma unroll
  for (int i = 0; i < 8; ++i) { c_reg[i] = 0.f; h_reg[i] = 0.f; }

  const v8f zero = {0.f, 0.f, 0.f, 0.f, 0.f, 0.f, 0.f, 0.f};

  for (int step = 0; step < TT; ++step) {
    const int t = dir ? (TT - 1 - step) : step;

    v8f acc[4];
#pragma unroll
    for (int g = 0; g < 4; ++g) acc[g] = zero;

    LFrag f0, f1;
    l_load(h_lds, whh, ln, half, col, 0, f0);
#pragma unroll 2
    for (int k0 = 0; k0 < HH; k0 += 64) {
      l_load(h_lds, whh, ln, half, col, k0 + 32, f1);
      l_mma(f0, acc);
      l_load(h_lds, whh, ln, half, col, (k0 + 64) & (HH - 1), f0); // wraps to dummy 0
      l_mma(f1, acc);
    }
    __syncthreads();   // all h_lds reads for this step done

    if (half == 0) {   // rows 0..7 are the real batch
#pragma unroll
      for (int i = 0; i < 8; ++i) {
        const float* gpr = gp + (size_t)(t * BB + i) * G4H;
        float ig = sigm(acc[0][i] + gpr[0 * HH + col]);
        float fg = sigm(acc[1][i] + gpr[1 * HH + col]);
        float gg = tanhf(acc[2][i] + gpr[2 * HH + col]);
        float og = sigm(acc[3][i] + gpr[3 * HH + col]);
        float c = fg * c_reg[i] + ig * gg;
        float h = og * tanhf(c);
        c_reg[i] = c;
        h_reg[i] = h;
        unsigned short hb = f2bf(h);
        h_lds[i * HH + col] = hb;
        out[(size_t)(t * BB + i) * HH + col] = hb;
      }
    }
    __syncthreads();   // h_lds updated before next step reads
  }

  if (dir == 0 && half == 0) {
#pragma unroll
    for (int i = 0; i < 8; ++i) {
      hT_dst[i * HH + col] = h_reg[i];
      cT_dst[i * HH + col] = c_reg[i];
    }
  }
}

// ---------------------------------------------------------------- launch
extern "C" void kernel_launch(void* const* d_in, const int* in_sizes, int n_in,
                              void* d_out, int out_size, void* d_ws, size_t ws_size,
                              hipStream_t stream) {
  (void)in_sizes; (void)n_in; (void)out_size; (void)ws_size;

  const int*   x      = (const int*)d_in[0];
  const float* embed  = (const float*)d_in[1];
  const float* fcW    = (const float*)d_in[2];
  const float* fcb    = (const float*)d_in[3];
  // order: fwd0, fwd1, bwd0, bwd1  ->  dl index 0..3
  const float* Wih_f[4] = {(const float*)d_in[4], (const float*)d_in[7],
                           (const float*)d_in[10], (const float*)d_in[13]};
  const float* bih_f[4] = {(const float*)d_in[5], (const float*)d_in[8],
                           (const float*)d_in[11], (const float*)d_in[14]};
  const float* Whh_f[4] = {(const float*)d_in[6], (const float*)d_in[9],
                           (const float*)d_in[12], (const float*)d_in[15]};

  char* ws = (char*)d_ws;
  size_t off = 0;
  auto carve = [&](size_t bytes) -> void* {
    void* p = ws + off;
    off = (off + bytes + 255) & ~(size_t)255;
    return p;
  };

  unsigned short* Xemb = (unsigned short*)carve((size_t)MROWS * EE * 2);
  unsigned short* WihB[4], *WhhB[4];
  for (int i = 0; i < 4; ++i) WihB[i] = (unsigned short*)carve((size_t)G4H * EE * 2);
  for (int i = 0; i < 4; ++i) WhhB[i] = (unsigned short*)carve((size_t)G4H * HH * 2);
  unsigned short* fcWB  = (unsigned short*)carve((size_t)VV * 2 * HH * 2);
  float* gatesF = (float*)carve((size_t)MROWS * G4H * 4);
  float* gatesB = (float*)carve((size_t)MROWS * G4H * 4);
  unsigned short* outs[4];
  for (int i = 0; i < 4; ++i) outs[i] = (unsigned short*)carve((size_t)MROWS * HH * 2);
  unsigned short* comb = (unsigned short*)carve((size_t)MROWS * 2 * HH * 2);

  float* logits = (float*)d_out;
  float* hT = logits + (size_t)BB * TT * VV;   // [L,B,H]
  float* cT = hT + 2 * BB * HH;                // [L,B,H]

  const int CVT = 256;
  const int wElems = G4H * EE;                 // 1,048,576 (Wih == Whh size here)
  for (int i = 0; i < 4; ++i) {
    cvt_bf16_kernel<<<(wElems + CVT - 1) / CVT, CVT, 0, stream>>>(Wih_f[i], WihB[i], wElems);
    cvt_bf16_kernel<<<(wElems + CVT - 1) / CVT, CVT, 0, stream>>>(Whh_f[i], WhhB[i], wElems);
  }
  const int fcElems = VV * 2 * HH;             // 32,768,000
  cvt_bf16_kernel<<<(fcElems + CVT - 1) / CVT, CVT, 0, stream>>>(fcW, fcWB, fcElems);

  const int embElems = TT * BB * EE;
  embed_kernel<<<(embElems + CVT - 1) / CVT, CVT, 0, stream>>>(x, embed, Xemb);

  dim3 gblk(32, 4);
  // layer 0 input projections
  gemm_bias_kernel<<<dim3(G4H / 256, MROWS / 32), gblk, 0, stream>>>(
      Xemb, WihB[0], bih_f[0], gatesF, MROWS, G4H, EE);
  gemm_bias_kernel<<<dim3(G4H / 256, MROWS / 32), gblk, 0, stream>>>(
      Xemb, WihB[2], bih_f[2], gatesB, MROWS, G4H, EE);
  // layer 0 recurrence (fwd + bwd concurrently)
  lstm_layer_kernel<<<2, dim3(32, 32), 0, stream>>>(
      gatesF, gatesB, WhhB[0], WhhB[2], outs[0], outs[2],
      hT + 0 * BB * HH, cT + 0 * BB * HH);
  // layer 1 input projections
  gemm_bias_kernel<<<dim3(G4H / 256, MROWS / 32), gblk, 0, stream>>>(
      outs[0], WihB[1], bih_f[1], gatesF, MROWS, G4H, HH);
  gemm_bias_kernel<<<dim3(G4H / 256, MROWS / 32), gblk, 0, stream>>>(
      outs[2], WihB[3], bih_f[3], gatesB, MROWS, G4H, HH);
  // layer 1 recurrence
  lstm_layer_kernel<<<2, dim3(32, 32), 0, stream>>>(
      gatesF, gatesB, WhhB[1], WhhB[3], outs[1], outs[3],
      hT + 1 * BB * HH, cT + 1 * BB * HH);
  // concat fwd/bwd -> [B*T, 2H]
  const int cElems = MROWS * HH;
  combine_kernel<<<(cElems + CVT - 1) / CVT, CVT, 0, stream>>>(outs[1], outs[3], comb);
  // final FC: [2048,1024] @ [1024,32000]
  gemm_bias_kernel<<<dim3(VV / 256, MROWS / 32), gblk, 0, stream>>>(
      comb, fcWB, fcb, logits, MROWS, VV, 2 * HH);
}
